// GraphConvBn_46986942218275
// MI455X (gfx1250) — compile-verified, matrix operand order checked by reading, not run on hardware
//
#include <hip/hip_runtime.h>

#define N_NODES 100000
#define N_EDGES 1600000
#define D 128           // D_IN == D_OUT == 128

typedef __attribute__((ext_vector_type(16))) __bf16        v16bf;
typedef __attribute__((ext_vector_type(8)))  float         v8f;
typedef __attribute__((ext_vector_type(8)))  unsigned int  v8u;

// Pack two f32 -> packed bf16x2 (round-to-nearest, ties-up) in 3 VALU ops:
// two v_add_u32 + one v_perm_b32 selecting the high halves.
__device__ __forceinline__ unsigned int pack_bf16(float lo, float hi) {
    unsigned int a = __builtin_bit_cast(unsigned int, lo) + 0x8000u;
    unsigned int b = __builtin_bit_cast(unsigned int, hi) + 0x8000u;
    return __builtin_amdgcn_perm(b, a, 0x07060302u);   // {b[31:16], a[31:16]}
}

// ---------------------------------------------------------------------------
// Zero aggr accumulator + BN stat accumulators (re-run every call: atomics).
// ---------------------------------------------------------------------------
__global__ __launch_bounds__(256) void zero_ws(float4* __restrict__ p, int n4) {
    int i = blockIdx.x * 256 + threadIdx.x;
    if (i < n4) p[i] = make_float4(0.f, 0.f, 0.f, 0.f);
}

// ---------------------------------------------------------------------------
// Scatter-add: one wave per edge; lane c owns channels [4c, 4c+3].
// x rows are 512B -> one wave reads exactly one row fully coalesced.
// aggr updates are native f32 atomics, L2-resident (x+aggr fit in 192MB L2).
// ---------------------------------------------------------------------------
__global__ __launch_bounds__(256) void scatter_edges(const float* __restrict__ x,
                                                     const long long* __restrict__ ei,
                                                     float* __restrict__ aggr) {
    int gid  = blockIdx.x * 256 + threadIdx.x;
    int e    = gid >> 5;
    int lane = gid & 31;
    if (e >= N_EDGES) return;
    long long s = ei[e];
    long long d = ei[N_EDGES + e];
    float4 v = ((const float4*)(x + (size_t)s * D))[lane];
    float* a = aggr + (size_t)d * D + lane * 4;
    unsafeAtomicAdd(a + 0, v.x);
    unsafeAtomicAdd(a + 1, v.y);
    unsafeAtomicAdd(a + 2, v.z);
    unsafeAtomicAdd(a + 3, v.w);
}

// ---------------------------------------------------------------------------
// Fused GEMM: h = [aggr | x] @ [W_rel ; W_root] + b_rel   via bf16 WMMA.
// Workgroup = 256 thr (8 waves) -> 128x128 output tile, K loop of 8 x 32.
// W (256x128) converted to bf16 and pre-swizzled into LDS so each lane's
// B fragment is one contiguous 32B ds_load. A fragments come straight from
// global (L2-resident) as 4x float4 per K-step. Epilogue writes h (aliased
// onto the aggr buffer) and accumulates per-channel sum/sumsq for BN.
//
// Fragment layouts (wave32, 16-bit):
//   A 16x32: lanes 0-15 -> M=lane, K = {0..7,16..23}; lanes 16-31 -> +8 in K.
//   B 32x16: lanes 0-15 -> N=lane, K = {0..7,16..23};        lanes 16-31 -> +8 in K.
// ---------------------------------------------------------------------------
__global__ __launch_bounds__(256) void gemm_bn_partial(
        const float* __restrict__ x,     const float* __restrict__ aggr,
        const float* __restrict__ Wroot, const float* __restrict__ Wrel,
        const float* __restrict__ brel,
        float* __restrict__ hbuf, float* __restrict__ sumbuf, float* __restrict__ sqbuf)
{
    __shared__ __align__(32) unsigned int ldsB[8 * 8 * 32 * 8];   // 64 KB

    const int tid = threadIdx.x;
    // ---- stage W = [W_rel ; W_root] as swizzled bf16 fragments -------------
    for (int i = tid; i < 8 * 8 * 32 * 8; i += 256) {
        int v    =  i        & 7;
        int lane = (i >> 3)  & 31;
        int nb   = (i >> 8)  & 7;
        int kb   =  i >> 11;
        int k    = kb * 32 + ((v & 4) ? 16 : 0) + ((lane & 16) ? 8 : 0) + (v & 3) * 2;
        int col  = nb * 16 + (lane & 15);
        const float* W = (k < D) ? (Wrel + (size_t)k * D + col)
                                 : (Wroot + (size_t)(k - D) * D + col);
        ldsB[i] = pack_bf16(W[0], W[D]);          // K pair (k, k+1)
    }
    __syncthreads();

    const int wave    = tid >> 5;
    const int lane    = tid & 31;
    const int rowBase = blockIdx.x * 128 + wave * 16;
    const int mrow    = rowBase + (lane & 15);
    const int mrowc   = (mrow < N_NODES) ? mrow : (N_NODES - 1);   // clamp tail
    const int kshift  = (lane & 16) ? 8 : 0;

    v8f acc[8] = {};

    #pragma unroll
    for (int kb = 0; kb < 8; ++kb) {
        const float* src = (kb < 4) ? (aggr + (size_t)mrowc * D + kb * 32)
                                    : (x    + (size_t)mrowc * D + (kb - 4) * 32);
        // Two contiguous 8-float runs per lane -> 4x 16B loads.
        float4 q0 = *(const float4*)(src + kshift);
        float4 q1 = *(const float4*)(src + kshift + 4);
        float4 q2 = *(const float4*)(src + 16 + kshift);
        float4 q3 = *(const float4*)(src + 16 + kshift + 4);
        v8u au;
        au[0] = pack_bf16(q0.x, q0.y);
        au[1] = pack_bf16(q0.z, q0.w);
        au[2] = pack_bf16(q1.x, q1.y);
        au[3] = pack_bf16(q1.z, q1.w);
        au[4] = pack_bf16(q2.x, q2.y);
        au[5] = pack_bf16(q2.z, q2.w);
        au[6] = pack_bf16(q3.x, q3.y);
        au[7] = pack_bf16(q3.z, q3.w);
        v16bf a = __builtin_bit_cast(v16bf, au);
        #pragma unroll
        for (int nb = 0; nb < 8; ++nb) {
            v8u bu = *(const v8u*)&ldsB[((kb * 8 + nb) * 32 + lane) * 8];
            v16bf b = __builtin_bit_cast(v16bf, bu);
            acc[nb] = __builtin_amdgcn_wmma_f32_16x16x32_bf16(
                          false, a, false, b, (short)0, acc[nb], false, false);
        }
    }

    // ---- epilogue: bias, store h, per-channel BN partial sums --------------
    const int colLane = lane & 15;
    const int rhalf   = (lane & 16) ? 8 : 0;
    const bool fullTile = (blockIdx.x * 128 + 128) <= N_NODES;   // uniform

    if (fullTile) {
        #pragma unroll
        for (int nb = 0; nb < 8; ++nb) {
            const int col  = nb * 16 + colLane;
            const float bias = brel[col];
            float s = 0.f, sq = 0.f;
            #pragma unroll
            for (int r = 0; r < 8; ++r) {
                int row = rowBase + rhalf + r;
                float h = acc[nb][r] + bias;
                hbuf[(size_t)row * D + col] = h;
                s  += h;
                sq += h * h;
            }
            unsafeAtomicAdd(&sumbuf[col], s);
            unsafeAtomicAdd(&sqbuf[col],  sq);
        }
    } else {
        #pragma unroll
        for (int nb = 0; nb < 8; ++nb) {
            const int col  = nb * 16 + colLane;
            const float bias = brel[col];
            float s = 0.f, sq = 0.f;
            #pragma unroll
            for (int r = 0; r < 8; ++r) {
                int row = rowBase + rhalf + r;
                if (row < N_NODES) {
                    float h = acc[nb][r] + bias;
                    hbuf[(size_t)row * D + col] = h;
                    s  += h;
                    sq += h * h;
                }
            }
            unsafeAtomicAdd(&sumbuf[col], s);
            unsafeAtomicAdd(&sqbuf[col],  sq);
        }
    }
}

// ---------------------------------------------------------------------------
// Fold sums into affine coefficients:  out = h*a + b ; a = gamma*rsqrt(var+eps)
// ---------------------------------------------------------------------------
__global__ void bn_finalize(const float* __restrict__ sumbuf, const float* __restrict__ sqbuf,
                            const float* __restrict__ gamma,  const float* __restrict__ beta,
                            float* __restrict__ coefA, float* __restrict__ coefB) {
    int c = threadIdx.x;
    float mean = sumbuf[c] * (1.0f / N_NODES);
    float var  = sqbuf[c]  * (1.0f / N_NODES) - mean * mean;
    float inv  = rsqrtf(var + 1e-5f);
    float a    = gamma[c] * inv;
    coefA[c] = a;
    coefB[c] = beta[c] - mean * a;
}

// ---------------------------------------------------------------------------
// Elementwise normalize + ReLU, float4-vectorized single pass.
// ---------------------------------------------------------------------------
__global__ __launch_bounds__(256) void bn_relu(const float* __restrict__ hbuf,
                                               const float* __restrict__ coefA,
                                               const float* __restrict__ coefB,
                                               float* __restrict__ out) {
    int i = blockIdx.x * 256 + threadIdx.x;         // float4 index
    const int n4 = N_NODES * D / 4;
    if (i >= n4) return;
    float4 h = ((const float4*)hbuf)[i];
    int c = (i * 4) & (D - 1);
    float4 o;
    o.x = fmaxf(0.f, h.x * coefA[c + 0] + coefB[c + 0]);
    o.y = fmaxf(0.f, h.y * coefA[c + 1] + coefB[c + 1]);
    o.z = fmaxf(0.f, h.z * coefA[c + 2] + coefB[c + 2]);
    o.w = fmaxf(0.f, h.w * coefA[c + 3] + coefB[c + 3]);
    ((float4*)out)[i] = o;
}

// ---------------------------------------------------------------------------
extern "C" void kernel_launch(void* const* d_in, const int* in_sizes, int n_in,
                              void* d_out, int out_size, void* d_ws, size_t ws_size,
                              hipStream_t stream) {
    (void)in_sizes; (void)n_in; (void)out_size; (void)ws_size;
    const float*     x     = (const float*)d_in[0];
    const long long* ei    = (const long long*)d_in[1];   // int64 edge_index [2, E]
    const float*     Wroot = (const float*)d_in[2];
    const float*     Wrel  = (const float*)d_in[3];
    const float*     brel  = (const float*)d_in[4];
    const float*     gamma = (const float*)d_in[5];
    const float*     beta  = (const float*)d_in[6];
    float*           out   = (float*)d_out;

    // workspace layout: aggr/h (aliased, N*D f32) | sum | sumsq | coefA | coefB
    float* aggr   = (float*)d_ws;
    float* sumbuf = aggr + (size_t)N_NODES * D;
    float* sqbuf  = sumbuf + D;
    float* coefA  = sqbuf + D;
    float* coefB  = coefA + D;

    const int n4 = (N_NODES * D + 2 * D) / 4;     // aggr + sum + sumsq
    zero_ws<<<(n4 + 255) / 256, 256, 0, stream>>>((float4*)d_ws, n4);

    scatter_edges<<<(N_EDGES * 32) / 256, 256, 0, stream>>>(x, ei, aggr);

    gemm_bn_partial<<<(N_NODES + 127) / 128, 256, 0, stream>>>(
        x, aggr, Wroot, Wrel, brel, aggr /* hbuf aliases aggr */, sumbuf, sqbuf);

    bn_finalize<<<1, D, 0, stream>>>(sumbuf, sqbuf, gamma, beta, coefA, coefB);

    bn_relu<<<(N_NODES * D / 4 + 255) / 256, 256, 0, stream>>>(aggr, coefA, coefB, out);
}
